// LinearQuant4bitDUASQ_12403865551687
// MI455X (gfx1250) — 8-bit matrix operand order repaired and checked by reading, compiled, not run on hardware
//
#include <hip/hip_runtime.h>
#include <hip/hip_bf16.h>
#include <stdint.h>

typedef int v8i __attribute__((ext_vector_type(8)));

#define M_TOK 4096
#define N_OUT 4096
#define K_DIM 4096
#define K_HALF 2048
#define ROWB  80

__global__ __launch_bounds__(256)
void quant_x_kernel(const float* __restrict__ x, unsigned char* __restrict__ xq,
                    float* __restrict__ sx, float* __restrict__ zx,
                    int* __restrict__ sumx1, int* __restrict__ sumx2) {
  const int m = blockIdx.x;
  const int t = threadIdx.x;
  const float4* px = (const float4*)(x + (size_t)m * K_DIM + t * 16);
  float4 f[4];
  f[0] = px[0]; f[1] = px[1]; f[2] = px[2]; f[3] = px[3];

  float mn = 3.402823466e38f, mx = -3.402823466e38f;
#pragma unroll
  for (int i = 0; i < 4; ++i) {
    mn = fminf(mn, fminf(fminf(f[i].x, f[i].y), fminf(f[i].z, f[i].w)));
    mx = fmaxf(mx, fmaxf(fmaxf(f[i].x, f[i].y), fmaxf(f[i].z, f[i].w)));
  }

  __shared__ float smn[256], smx[256];
  __shared__ int ssum[256];
  smn[t] = mn; smx[t] = mx;
  __syncthreads();
  for (int s = 128; s > 0; s >>= 1) {
    if (t < s) {
      smn[t] = fminf(smn[t], smn[t + s]);
      smx[t] = fmaxf(smx[t], smx[t + s]);
    }
    __syncthreads();
  }
  const float gmn = smn[0], gmx = smx[0];
  const float scale = fmaxf((gmx - gmn) * (1.0f / 15.0f), 1e-8f);
  const float zero = rintf(-gmn / scale);

  int lsum = 0;
  unsigned int ow[4];
#pragma unroll
  for (int i = 0; i < 4; ++i) {
    float cc[4] = {f[i].x, f[i].y, f[i].z, f[i].w};
    unsigned int wbits = 0u;
#pragma unroll
    for (int j = 0; j < 4; ++j) {
      float q = rintf(cc[j] / scale) + zero;
      q = fminf(fmaxf(q, 0.0f), 15.0f);
      int qi = (int)q;
      lsum += qi;
      wbits |= ((unsigned int)qi) << (8 * j);
    }
    ow[i] = wbits;
  }
  *(uint4*)(xq + (size_t)m * K_DIM + t * 16) = make_uint4(ow[0], ow[1], ow[2], ow[3]);

  ssum[t] = lsum;
  __syncthreads();
  for (int s = 64; s > 0; s >>= 1) {
    if ((t & 127) < s) ssum[t] += ssum[t + s];
    __syncthreads();
  }
  if (t == 0)  { sumx1[m] = ssum[0];   sx[m] = scale; zx[m] = zero; }
  if (t == 128){ sumx2[m] = ssum[128]; }
}

__global__ __launch_bounds__(256)
void pack_w_kernel(const int* __restrict__ wqi, unsigned char* __restrict__ wq8,
                   int* __restrict__ sumw1, int* __restrict__ sumw2) {
  const int n = blockIdx.x;
  const int t = threadIdx.x;
  const int4* pw = (const int4*)(wqi + (size_t)n * K_DIM + t * 16);
  int4 wv[4];
  wv[0] = pw[0]; wv[1] = pw[1]; wv[2] = pw[2]; wv[3] = pw[3];

  int lsum = 0;
  unsigned int ow[4];
#pragma unroll
  for (int i = 0; i < 4; ++i) {
    lsum += wv[i].x + wv[i].y + wv[i].z + wv[i].w;
    ow[i] = (unsigned int)(wv[i].x & 0xFF)        |
            ((unsigned int)(wv[i].y & 0xFF) << 8) |
            ((unsigned int)(wv[i].z & 0xFF) << 16)|
            ((unsigned int)(wv[i].w & 0xFF) << 24);
  }
  *(uint4*)(wq8 + (size_t)n * K_DIM + t * 16) = make_uint4(ow[0], ow[1], ow[2], ow[3]);

  __shared__ int ssum[256];
  ssum[t] = lsum;
  __syncthreads();
  for (int s = 64; s > 0; s >>= 1) {
    if ((t & 127) < s) ssum[t] += ssum[t + s];
    __syncthreads();
  }
  if (t == 0)   sumw1[n] = ssum[0];
  if (t == 128) sumw2[n] = ssum[128];
}


#define ASYNC_CP16(ldsOff, gptr)                                               \
  asm volatile("global_load_async_to_lds_b128 %0, %1, off"                     \
               :: "v"((unsigned)(ldsOff)),                                     \
                  "v"((unsigned long long)(size_t)(gptr))                      \
               : "memory")

#define ASYNC_WAIT0() asm volatile("s_wait_asynccnt 0x0" ::: "memory")

template <int REGION>
__global__ __launch_bounds__(256)
void gemm_region_kernel(const unsigned char* __restrict__ xq,
                        const unsigned char* __restrict__ wq8,
                        const float* __restrict__ sx, const float* __restrict__ zx,
                        const int* __restrict__ sumx,
                        const int* __restrict__ sumw,
                        const float* __restrict__ wsc,
                        const float* __restrict__ wzp,
                        float* __restrict__ y) {
  __shared__ unsigned char As[2][128 * ROWB];
  __shared__ unsigned char Bs[2][128 * ROWB];

  const int tid   = threadIdx.x;
  const int lane  = tid & 31;
  const int wave  = tid >> 5;
  const int waveM = wave >> 2;
  const int waveN = wave & 3;
  const int mBase = blockIdx.y * 128;
  const int nBase = blockIdx.x * 128;
  const int llo = lane & 15;
  const int lhi = lane >> 4;
  const int kbase = REGION * K_HALF;

  const int c0 = tid, c1 = tid + 256;
  const int r0 = c0 >> 2, kb0 = (c0 & 3) * 16;
  const int r1 = c1 >> 2, kb1 = (c1 & 3) * 16;
  const unsigned char* gA0 = xq  + (size_t)(mBase + r0) * K_DIM + kbase + kb0;
  const unsigned char* gA1 = xq  + (size_t)(mBase + r1) * K_DIM + kbase + kb1;
  const unsigned char* gB0 = wq8 + (size_t)(nBase + r0) * K_DIM + kbase + kb0;
  const unsigned char* gB1 = wq8 + (size_t)(nBase + r1) * K_DIM + kbase + kb1;
  const unsigned aL0 = (unsigned)(size_t)&As[0][r0 * ROWB + kb0];
  const unsigned aL1 = (unsigned)(size_t)&As[0][r1 * ROWB + kb1];
  const unsigned bL0 = (unsigned)(size_t)&Bs[0][r0 * ROWB + kb0];
  const unsigned bL1 = (unsigned)(size_t)&Bs[0][r1 * ROWB + kb1];

  v8i acc[4][2];
#pragma unroll
  for (int tt = 0; tt < 4; ++tt)
#pragma unroll
    for (int u = 0; u < 2; ++u) {
      v8i z = {0, 0, 0, 0, 0, 0, 0, 0};
      acc[tt][u] = z;
    }

  ASYNC_CP16(aL0, gA0);
  ASYNC_CP16(aL1, gA1);
  ASYNC_CP16(bL0, gB0);
  ASYNC_CP16(bL1, gB1);
  ASYNC_WAIT0();
  __syncthreads();

#pragma unroll
  for (int ks = 0; ks < 32; ++ks) {
    const int buf = ks & 1;

    if (ks + 1 < 32) {
      const int k0 = (ks + 1) * 64;
      const unsigned d = (unsigned)(buf ^ 1) * (128 * ROWB);
      ASYNC_CP16(aL0 + d, gA0 + k0);
      ASYNC_CP16(aL1 + d, gA1 + k0);
      ASYNC_CP16(bL0 + d, gB0 + k0);
      ASYNC_CP16(bL1 + d, gB1 + k0);
    }

    v8i af[4];
#pragma unroll
    for (int tt = 0; tt < 4; ++tt) {
      const int r  = waveM * 64 + tt * 16 + llo;
      const int ko = lhi * 8;
      const unsigned char* p = &As[buf][r * ROWB + ko];
      uint2 d0 = *(const uint2*)(p);
      uint2 d1 = *(const uint2*)(p + 16);
      uint2 d2 = *(const uint2*)(p + 32);
      uint2 d3 = *(const uint2*)(p + 48);
      v8i a = {(int)d0.x, (int)d0.y, (int)d1.x, (int)d1.y,
               (int)d2.x, (int)d2.y, (int)d3.x, (int)d3.y};
      af[tt] = a;
    }

    v8i bf[2];
#pragma unroll
    for (int u = 0; u < 2; ++u) {
      const int n  = waveN * 32 + u * 16 + llo;
      const int ko = lhi * 8;
      const unsigned char* p = &Bs[buf][n * ROWB + ko];
      uint2 e0 = *(const uint2*)(p);
      uint2 e1 = *(const uint2*)(p + 16);
      uint2 e2 = *(const uint2*)(p + 32);
      uint2 e3 = *(const uint2*)(p + 48);
      v8i b = {(int)e0.x, (int)e0.y, (int)e1.x, (int)e1.y,
               (int)e2.x, (int)e2.y, (int)e3.x, (int)e3.y};
      bf[u] = b;
    }

#pragma unroll
    for (int tt = 0; tt < 4; ++tt)
#pragma unroll
      for (int u = 0; u < 2; ++u)
        acc[tt][u] = __builtin_amdgcn_wmma_i32_16x16x64_iu8(
            false, af[tt], false, bf[u], acc[tt][u], false, false);

    ASYNC_WAIT0();
    __syncthreads();
  }

  float swv[2], zwv[2], swsum[2];
#pragma unroll
  for (int u = 0; u < 2; ++u) {
    const int n = nBase + waveN * 32 + u * 16 + llo;
    swv[u] = wsc[n]; zwv[u] = wzp[n]; swsum[u] = (float)sumw[n];
  }
#pragma unroll
  for (int tt = 0; tt < 4; ++tt) {
#pragma unroll
    for (int v = 0; v < 8; ++v) {
      const int m = mBase + waveM * 64 + tt * 16 + lhi * 8 + v;
      const float zxv = zx[m];
      const float sxr = (float)sumx[m];
#pragma unroll
      for (int u = 0; u < 2; ++u) {
        const int n = nBase + waveN * 32 + u * 16 + llo;
        const float a = (float)acc[tt][u][v];
        const float tterm =
            a - zxv * swsum[u] - zwv[u] * sxr + 2048.0f * zxv * zwv[u];
        const size_t idx = (size_t)m * N_OUT + n;
        if (REGION == 0) {
          y[idx] = swv[u] * tterm;
        } else {
          y[idx] = sx[m] * (y[idx] + swv[u] * tterm);
        }
      }
    }
  }
}

extern "C" void kernel_launch(void* const* d_in, const int* in_sizes, int n_in,
                              void* d_out, int out_size, void* d_ws, size_t ws_size,
                              hipStream_t stream) {
  (void)in_sizes; (void)n_in; (void)out_size; (void)ws_size;
  const float* x   = (const float*)d_in[0];
  const int*   w_q = (const int*)d_in[1];
  const float* ws1 = (const float*)d_in[2];
  const float* wz1 = (const float*)d_in[3];
  const float* ws2 = (const float*)d_in[4];
  const float* wz2 = (const float*)d_in[5];
  float* y = (float*)d_out;

  unsigned char* ws  = (unsigned char*)d_ws;
  unsigned char* xq  = ws;
  unsigned char* wq8 = ws + ((size_t)16 << 20);
  unsigned char* p   = ws + ((size_t)32 << 20);
  float* sx    = (float*)(p + 0 * 16384);
  float* zx    = (float*)(p + 1 * 16384);
  int*   sumx1 = (int*)  (p + 2 * 16384);
  int*   sumx2 = (int*)  (p + 3 * 16384);
  int*   sumw1 = (int*)  (p + 4 * 16384);
  int*   sumw2 = (int*)  (p + 5 * 16384);

  quant_x_kernel<<<M_TOK, 256, 0, stream>>>(x, xq, sx, zx, sumx1, sumx2);
  pack_w_kernel<<<N_OUT, 256, 0, stream>>>(w_q, wq8, sumw1, sumw2);
  gemm_region_kernel<0><<<dim3(N_OUT / 128, M_TOK / 128), 256, 0, stream>>>(
      xq, wq8, sx, zx, sumx1, sumw1, ws1, wz1, y);
  gemm_region_kernel<1><<<dim3(N_OUT / 128, M_TOK / 128), 256, 0, stream>>>(
      xq, wq8, sx, zx, sumx2, sumw2, ws2, wz2, y);
}
